// SparseMLP_678604832884
// MI455X (gfx1250) — matrix operand order repaired and checked by reading, not compiled, not run on hardware
//
#include <hip/hip_runtime.h>
#include <hip/hip_bf16.h>
#include <stdint.h>

// ---------------------------------------------------------------------------
// SparseMLP for MI455X (gfx1250, wave32, WMMA).
//   h = gelu(gather(x, perm) @ w_up + b_up) @ w_down + b_down, scattered back.
// 137 GFLOP; bf16 v_wmma_f32_16x16x32_bf16 path (fp32 WMMA only has K=4).
// Fully fused: the 512 MB fp32 intermediate never exists; it lives in LDS as
// bf16 one K-half at a time. GELU uses the CDNA5 hardware tanh transcendental
// so it co-executes under the XDL/WMMA shadow instead of a branchy erff libm
// expansion.
// ---------------------------------------------------------------------------

#define HEADS    16
#define HEAD_IN  256
#define HEAD_OUT 1024
#define FULLD    4096
#define BROWS    32      // rows per workgroup (2 m-tiles of 16)

typedef __attribute__((ext_vector_type(16))) __bf16 v16bf;
typedef __attribute__((ext_vector_type(8)))  float  v8f;

union FragU { v16bf v; uint4 q[2]; };

__device__ __forceinline__ unsigned short bfbits(float f) {
    union { float f; uint32_t u; } v; v.f = f;
    uint32_t r = v.u + 0x7FFFu + ((v.u >> 16) & 1u);   // round-to-nearest-even
    return (unsigned short)(r >> 16);
}

__device__ __forceinline__ __bf16 f2bf(float f) {
    union { unsigned short u; __bf16 b; } o; o.u = bfbits(f);
    return o.b;
}

// CDNA5 hardware tanh (TRANS pipe) — branch-free, co-executes with WMMA.
__device__ __forceinline__ float htanh(float x) {
#if __has_builtin(__builtin_amdgcn_tanhf)
    return __builtin_amdgcn_tanhf(x);
#elif __has_builtin(__builtin_amdgcn_tanh_f32)
    return __builtin_amdgcn_tanh_f32(x);
#else
    float r; asm("v_tanh_f32 %0, %1" : "=v"(r) : "v"(x)); return r;
#endif
}

// tanh-form GELU: 5 VALU + 1 TRANS per element, no divergence.
__device__ __forceinline__ float gelu_fast(float x) {
    float x2 = x * x;
    float p  = __builtin_fmaf(x2, 0.0356774081f, 0.79788456080286536f);
    float t  = htanh(x * p);
    float hx = 0.5f * x;
    return __builtin_fmaf(hx, t, hx);
}

// A-fragment (16x32 bf16) from an LDS row-major tile.
// Lane L supplies row M=L&15.  Layout: element i -> K = k0 + (L>>4)*16 + i
// (matches B swizzle)  => two contiguous 16B LDS reads.
__device__ __forceinline__ v16bf ldsA(const __bf16* rowBase, int k0, int half) {
    FragU f;
    const uint4* p = reinterpret_cast<const uint4*>(rowBase + k0 + half * 16);
    f.q[0] = p[0];
    f.q[1] = p[1];            // +8 elements = +16 bytes
    return f.v;
}

// B-fragment: pre-swizzled in global memory, lane L reads 32 contiguous bytes.
__device__ __forceinline__ v16bf ldB(const __bf16* p) {
    FragU f;
    const uint4* q = reinterpret_cast<const uint4*>(p);
    f.q[0] = q[0];
    f.q[1] = q[1];
    return f.v;
}

// ---------------------------------------------------------------------------
// Prep: fp32 weights -> bf16, laid out in WMMA B-fragment order.
// Block of a B fragment (K-tile 32 x N-tile 16): lane L holds 16 bf16 where
// element i = W[K = kt*32 + (L>>4)*16 + i][N = nt*16 + (L&15)].
// w_up  swizzle: block id = ((h*64 + nt)*8  + kt),  nt in [0,64), kt in [0,8)
// w_down swizzle: block id = ((h*16 + nt)*32 + kt), nt in [0,16), kt in [0,32)
// Weights (16 MB bf16 total) then stay resident in the 192 MB L2.
// ---------------------------------------------------------------------------
__global__ __launch_bounds__(256) void swizzle_weights(
    const float* __restrict__ wup, const float* __restrict__ wdn,
    __bf16* __restrict__ su, __bf16* __restrict__ sd) {
    int t = blockIdx.x * 256 + threadIdx.x;          // 0 .. 524287
    if (t < 262144) {                                // w_up: 16*64*8 blocks * 32 lanes
        int L  = t & 31;  int b  = t >> 5;
        int kt = b & 7;   int nb = b >> 3;
        int nt = nb & 63; int h  = nb >> 6;
        int col   = nt * 16 + (L & 15);
        int krow0 = kt * 32 + (L >> 4) * 16;
        const float* src = wup + ((size_t)(h * HEAD_IN + krow0)) * HEAD_OUT + col;
        __bf16* dst = su + (size_t)t * 16;
        #pragma unroll
        for (int i = 0; i < 16; ++i) dst[i] = f2bf(src[(size_t)i * HEAD_OUT]);
    } else {                                         // w_down: 16*16*32 blocks * 32 lanes
        int t2 = t - 262144;
        int L  = t2 & 31; int b  = t2 >> 5;
        int kt = b & 31;  int nb = b >> 5;
        int nt = nb & 15; int h  = nb >> 4;
        int col   = nt * 16 + (L & 15);
        int krow0 = kt * 32 + (L >> 4) * 16;
        const float* src = wdn + ((size_t)(h * HEAD_OUT + krow0)) * HEAD_IN + col;
        __bf16* dst = sd + (size_t)t2 * 16;
        #pragma unroll
        for (int i = 0; i < 16; ++i) dst[i] = f2bf(src[(size_t)i * HEAD_IN]);
    }
}

// ---------------------------------------------------------------------------
// Fused MLP. One block = 32 rows x 1 head, 8 waves (wave32).
// HEAD_OUT split in two K-halves of 512 so the GELU intermediate fits in LDS.
// ---------------------------------------------------------------------------
__global__ __launch_bounds__(256) void sparse_mlp(
    const float* __restrict__ x, const int* __restrict__ perm,
    const __bf16* __restrict__ wsu, const __bf16* __restrict__ wsd,
    const float* __restrict__ bup, const float* __restrict__ bdn,
    float* __restrict__ out) {

    __shared__ __align__(16) __bf16 lds_x[BROWS * HEAD_IN];   // 16 KB
    __shared__ __align__(16) __bf16 lds_h[BROWS * 512];       // 32 KB (one K-half)

    const int h       = blockIdx.x & (HEADS - 1);
    const int rowBase = (blockIdx.x >> 4) * BROWS;
    const int t       = threadIdx.x;
    const int lane    = t & 31;
    const int w       = t >> 5;          // wave id 0..7
    const int half    = lane >> 4;
    const int l16     = lane & 15;

    // ---- stage chunk-gathered x tile as bf16 ------------------------------
    // perm is contiguous in 64-wide chunks, and chunk bases are 64-element
    // aligned, so a float4 load at a 4-aligned column never crosses a chunk.
    {
        const int cg = (t & 63) * 4;                 // column base 0..252
        const int rr = t >> 6;                       // 0..3
        const int gsrc = perm[h * HEAD_IN + cg];     // +1..+3 are contiguous
        #pragma unroll
        for (int it = 0; it < 8; ++it) {
            const int m = it * 4 + rr;
            const float4 v4 = *reinterpret_cast<const float4*>(
                x + (size_t)(rowBase + m) * FULLD + gsrc);
            ushort4 o;
            o.x = bfbits(v4.x); o.y = bfbits(v4.y);
            o.z = bfbits(v4.z); o.w = bfbits(v4.w);
            *reinterpret_cast<ushort4*>(&lds_x[m * HEAD_IN + cg]) = o;
        }
    }
    __syncthreads();

    v8f dacc[2][2];                               // [m-tile][n'-tile] fp32 accum
    dacc[0][0] = v8f{}; dacc[0][1] = v8f{};
    dacc[1][0] = v8f{}; dacc[1][1] = v8f{};

    const __bf16* lrow0 = lds_x + l16 * HEAD_IN;        // A rows for m-tile 0
    const __bf16* lrow1 = lds_x + (16 + l16) * HEAD_IN; // A rows for m-tile 1
    const __bf16* hrow0 = lds_h + l16 * 512;
    const __bf16* hrow1 = lds_h + (16 + l16) * 512;

    for (int hk = 0; hk < 2; ++hk) {              // K-half of the 1024 hidden dim
        // ---------------- UP:  x(32x256) @ w_up -> gelu -> lds_h (32x512) ----
        #pragma unroll
        for (int nt = 0; nt < 4; ++nt) {
            const int nt_g = hk * 32 + w * 4 + nt;        // global 16-col tile
            v8f acc0 = v8f{}, acc1 = v8f{};
            const __bf16* bptr =
                wsu + ((size_t)((h * 64 + nt_g) * 8) * 32 + lane) * 16;
            #pragma unroll
            for (int kt = 0; kt < 8; ++kt) {
                v16bf bF = ldB(bptr + (size_t)kt * 512);
                v16bf a0 = ldsA(lrow0, kt * 32, half);
                acc0 = __builtin_amdgcn_wmma_f32_16x16x32_bf16(
                           false, a0, false, bF, (short)0, acc0, false, false);
                v16bf a1 = ldsA(lrow1, kt * 32, half);
                acc1 = __builtin_amdgcn_wmma_f32_16x16x32_bf16(
                           false, a1, false, bF, (short)0, acc1, false, false);
            }
            const float bias = bup[h * HEAD_OUT + nt_g * 16 + l16];
            const int   ncol = (w * 4 + nt) * 16 + l16;   // col in this K-half
            #pragma unroll
            for (int r = 0; r < 8; ++r) {                 // C layout: M=r+8*half
                const int M0 = r + 8 * half;
                lds_h[M0 * 512 + ncol]        = f2bf(gelu_fast(acc0[r] + bias));
                lds_h[(16 + M0) * 512 + ncol] = f2bf(gelu_fast(acc1[r] + bias));
            }
        }
        __syncthreads();

        // ---------------- DOWN: lds_h(32x512) @ w_down -> accumulate ---------
        {
            const __bf16* d0 =
                wsd + ((size_t)((h * 16 + w * 2 + 0) * 32) * 32 + lane) * 16;
            const __bf16* d1 =
                wsd + ((size_t)((h * 16 + w * 2 + 1) * 32) * 32 + lane) * 16;
            #pragma unroll 4
            for (int ktl = 0; ktl < 16; ++ktl) {
                const int ktg = hk * 16 + ktl;            // global K-tile of 32
                v16bf b0 = ldB(d0 + (size_t)ktg * 512);
                v16bf b1 = ldB(d1 + (size_t)ktg * 512);
                v16bf a0 = ldsA(hrow0, ktl * 32, half);
                v16bf a1 = ldsA(hrow1, ktl * 32, half);
                dacc[0][0] = __builtin_amdgcn_wmma_f32_16x16x32_bf16(
                                 false, a0, false, b0, (short)0, dacc[0][0], false, false);
                dacc[0][1] = __builtin_amdgcn_wmma_f32_16x16x32_bf16(
                                 false, a0, false, b1, (short)0, dacc[0][1], false, false);
                dacc[1][0] = __builtin_amdgcn_wmma_f32_16x16x32_bf16(
                                 false, a1, false, b0, (short)0, dacc[1][0], false, false);
                dacc[1][1] = __builtin_amdgcn_wmma_f32_16x16x32_bf16(
                                 false, a1, false, b1, (short)0, dacc[1][1], false, false);
            }
        }
        __syncthreads();          // lds_h reused by next K-half's UP phase
    }

    // ---- epilogue: + b_down, chunk-contiguous scatter through perm ----------
    #pragma unroll
    for (int ntp = 0; ntp < 2; ++ntp) {
        const int nloc = (w * 2 + ntp) * 16 + l16;        // 0..255 in head
        const int gcol = perm[h * HEAD_IN + nloc];
        const float bias = bdn[h * HEAD_IN + nloc];
        #pragma unroll
        for (int mt = 0; mt < 2; ++mt) {
            #pragma unroll
            for (int r = 0; r < 8; ++r) {
                const int row = rowBase + mt * 16 + r + 8 * half;
                out[(size_t)row * FULLD + gcol] = dacc[mt][ntp][r] + bias;
            }
        }
    }
}

// ---------------------------------------------------------------------------
extern "C" void kernel_launch(void* const* d_in, const int* in_sizes, int n_in,
                              void* d_out, int out_size, void* d_ws, size_t ws_size,
                              hipStream_t stream) {
    const float* x    = (const float*)d_in[0];
    const int*   perm = (const int*)  d_in[1];
    const float* wup  = (const float*)d_in[2];
    const float* bup  = (const float*)d_in[3];
    const float* wdn  = (const float*)d_in[4];
    const float* bdn  = (const float*)d_in[5];
    float*       out  = (float*)d_out;

    __bf16* su = (__bf16*)d_ws;                 // 4,194,304 bf16 = 8 MB
    __bf16* sd = su + 4194304;                  // 8 MB more

    // weight bf16 swizzle: 524288 threads
    swizzle_weights<<<2048, 256, 0, stream>>>(wup, wdn, su, sd);

    // fused MLP: (8192/32) row-blocks * 16 heads = 4096 blocks, 8 waves each
    sparse_mlp<<<4096, 256, 0, stream>>>(x, perm, su, sd, bup, bdn, out);
}
